// SwinTransformerBlock3D_5884105195634
// MI455X (gfx1250) — matrix operand order repaired and checked by reading, not compiled, not run on hardware
//
#include <hip/hip_runtime.h>

typedef __attribute__((ext_vector_type(16))) _Float16 v16h;
typedef __attribute__((ext_vector_type(8)))  float    v8f;

#define HSTR 72    // hbuf stride (halves)  : 144B rows, 16B aligned, bank-clean
#define ASTR 72    // bufA/B/C stride (halves)
#define SSTR 68    // sbuf stride (floats)  : 272B rows, 16B aligned
#define GSTR 264   // gbuf stride (halves)  : 528B rows, 16B aligned

__device__ __forceinline__ v8f wmma_f16(v16h a, v16h b, v8f c) {
    return __builtin_amdgcn_wmma_f32_16x16x32_f16(false, a, false, b, (short)0, c, false, false);
}

__device__ __forceinline__ float red4_add(float v) {
    v += __shfl_xor(v, 1, 32);
    v += __shfl_xor(v, 2, 32);
    return v;
}
__device__ __forceinline__ float red4_max(float v) {
    v = fmaxf(v, __shfl_xor(v, 1, 32));
    v = fmaxf(v, __shfl_xor(v, 2, 32));
    return v;
}

// A fragment (16x32 f16) from row-major buffer: row m, contiguous k
__device__ __forceinline__ v16h ldA(const _Float16* base, int stride, int m0, int k0, int lane) {
    const _Float16* p = base + (m0 + (lane & 15)) * stride + k0 + ((lane >> 4) << 4);
    v16h a;
#pragma unroll
    for (int i = 0; i < 16; ++i) a[i] = p[i];
    return a;
}

// B fragment (32x16 f16) from row-major B^T buffer (row = N, cols = K)
__device__ __forceinline__ v16h ldB(const _Float16* base, int stride, int n0, int k0, int lane) {
    const _Float16* p = base + (n0 + (lane & 15)) * stride + k0 + ((lane >> 4) << 4);
    v16h b;
#pragma unroll
    for (int i = 0; i < 16; ++i) b[i] = p[i];
    return b;
}

// D (16x16 f32) -> f16 LDS, val = (d + bias[n]) * scale
__device__ __forceinline__ void stD16(_Float16* base, int stride, int m0, int n0, int lane,
                                      v8f d, float scale, const float* __restrict__ bias) {
    int col = n0 + (lane & 15);
    float bb = bias[col];
    int mb = m0 + ((lane >> 4) << 3);
#pragma unroll
    for (int r = 0; r < 8; ++r) base[(mb + r) * stride + col] = (_Float16)((d[r] + bb) * scale);
}

// D -> f16 LDS, plain convert (no bias, no scale)
__device__ __forceinline__ void stD16p(_Float16* base, int stride, int m0, int n0, int lane, v8f d) {
    int col = n0 + (lane & 15);
    int mb = m0 + ((lane >> 4) << 3);
#pragma unroll
    for (int r = 0; r < 8; ++r) base[(mb + r) * stride + col] = (_Float16)d[r];
}

// D -> f16 LDS transposed (store element (m,n) at [n][m]), val = d + bias[n]
__device__ __forceinline__ void stD16T(_Float16* base, int stride, int m0, int n0, int lane,
                                       v8f d, const float* __restrict__ bias) {
    int col = n0 + (lane & 15);
    float bb = bias[col];
    int mb = m0 + ((lane >> 4) << 3);
#pragma unroll
    for (int r = 0; r < 8; ++r) base[col * stride + (mb + r)] = (_Float16)(d[r] + bb);
}

// D -> f32 LDS
__device__ __forceinline__ void stD32(float* base, int stride, int m0, int n0, int lane, v8f d) {
    int col = n0 + (lane & 15);
    int mb = m0 + ((lane >> 4) << 3);
#pragma unroll
    for (int r = 0; r < 8; ++r) base[(mb + r) * stride + col] = d[r];
}

// ---- weight prep: f32 -> f16, pre-transposed B^T layouts in workspace ----
// ws halves: [0)      wsq  [n=h*64+d][c]   16384
//            [16384)  wsk                   16384
//            [32768)  wsv                   16384
//            [49152)  wso  [c][kk=h*64+d]  16384
//            [65536)  wsw1 [f][c]          16384
//            [81920)  wsw2 [c][f]          16384
__global__ void prep_weights(const float* __restrict__ wq, const float* __restrict__ wk,
                             const float* __restrict__ wv, const float* __restrict__ wo,
                             const float* __restrict__ w1, const float* __restrict__ w2,
                             _Float16* __restrict__ ws) {
    int i = blockIdx.x * blockDim.x + threadIdx.x;   // 0 .. 98303
    if (i < 16384)      { int n = i >> 6, k = i & 63;                 ws[i] = (_Float16)wq[k * 256 + n]; }
    else if (i < 32768) { int j = i - 16384; int n = j >> 6, k = j & 63; ws[i] = (_Float16)wk[k * 256 + n]; }
    else if (i < 49152) { int j = i - 32768; int n = j >> 6, k = j & 63; ws[i] = (_Float16)wv[k * 256 + n]; }
    else if (i < 65536) { int j = i - 49152; int c = j >> 8, kk = j & 255; ws[i] = (_Float16)wo[kk * 64 + c]; }
    else if (i < 81920) { int j = i - 65536; int f = j >> 6, c = j & 63;   ws[i] = (_Float16)w1[c * 256 + f]; }
    else if (i < 98304) { int j = i - 81920; int c = j >> 8, f = j & 255;  ws[i] = (_Float16)w2[f * 64 + c]; }
}

// ---- fused Swin block: one 256-thread block per 4x4x4 window ----
__global__ __launch_bounds__(256) void swin_block(
    const float* __restrict__ x,
    const float* __restrict__ ln1_g, const float* __restrict__ ln1_b,
    const float* __restrict__ bq, const float* __restrict__ bk, const float* __restrict__ bv,
    const float* __restrict__ bo,
    const float* __restrict__ ln2_g, const float* __restrict__ ln2_b,
    const float* __restrict__ b1, const float* __restrict__ b2,
    const _Float16* __restrict__ ws, float* __restrict__ out)
{
    const int tid  = threadIdx.x;
    const int lane = tid & 31;
    const int wave = tid >> 5;
    const int win  = blockIdx.x;                       // 4096 windows
    const int wd = win >> 8, wh = (win >> 4) & 15, wwi = win & 15;
    const int baseVox = ((wd * 4) * 64 + wh * 4) * 64 + wwi * 4;

    __shared__ __align__(16) unsigned char smem[9216 + 45056];
    _Float16* hbuf = (_Float16*)smem;                    // 64 x HSTR halves (9216 B)
    unsigned char* u = smem + 9216;
    _Float16* bufA = (_Float16*)u;                       // q / P          (9216 B)
    _Float16* bufB = (_Float16*)(u + 9216);              // k / v^T        (9216 B)
    _Float16* bufC = (_Float16*)(u + 18432);             // per-head oh    (9216 B)
    float*    sbuf = (float*)(u + 27648);                // scores / y     (17408 B)
    _Float16* gbuf = (_Float16*)u;                       // FFN1 acts, aliases A/B/C/s (33792 B)

    // ---- LN1 + window gather: 4 lanes per token, 16 channels each ----
    {
        int m = tid >> 2, p = tid & 3;
        int vox = baseVox + (m >> 4) * 4096 + ((m >> 2) & 3) * 64 + (m & 3);
        const float4* xr = (const float4*)(x + (size_t)vox * 64 + p * 16);
        float4 qa[4];
#pragma unroll
        for (int j = 0; j < 4; ++j) qa[j] = xr[j];
        float s = 0.f, s2 = 0.f;
#pragma unroll
        for (int j = 0; j < 4; ++j) {
            s  += qa[j].x + qa[j].y + qa[j].z + qa[j].w;
            s2 += qa[j].x * qa[j].x + qa[j].y * qa[j].y + qa[j].z * qa[j].z + qa[j].w * qa[j].w;
        }
        s = red4_add(s); s2 = red4_add(s2);
        float mean = s * (1.f / 64.f);
        float var  = s2 * (1.f / 64.f) - mean * mean;
        float rstd = rsqrtf(var + 1e-5f);
        const float4* g4 = (const float4*)(ln1_g + p * 16);
        const float4* b4 = (const float4*)(ln1_b + p * 16);
        _Float16* hp = hbuf + m * HSTR + p * 16;
#pragma unroll
        for (int j = 0; j < 4; ++j) {
            float4 g = g4[j], b = b4[j], v = qa[j];
            hp[4 * j + 0] = (_Float16)((v.x - mean) * rstd * g.x + b.x);
            hp[4 * j + 1] = (_Float16)((v.y - mean) * rstd * g.y + b.y);
            hp[4 * j + 2] = (_Float16)((v.z - mean) * rstd * g.z + b.z);
            hp[4 * j + 3] = (_Float16)((v.w - mean) * rstd * g.w + b.w);
        }
    }
    __syncthreads();

    const _Float16* wsq  = ws;
    const _Float16* wsk  = ws + 16384;
    const _Float16* wsv  = ws + 32768;
    const _Float16* wso  = ws + 49152;
    const _Float16* wsw1 = ws + 65536;
    const _Float16* wsw2 = ws + 81920;

    // persistent out-projection accumulators: y is 64x64 = 16 tiles, 2 per wave
    v8f yacc0 = {}; v8f yacc1 = {};
    const int yt0 = wave * 2, yt1 = wave * 2 + 1;

    for (int hd = 0; hd < 4; ++hd) {
        const _Float16* wq_h = wsq + hd * 4096;
        const _Float16* wk_h = wsk + hd * 4096;
        const _Float16* wv_h = wsv + hd * 4096;

        // ---- Q & K projection (each 64x64, K=64) ----
#pragma unroll
        for (int t = 0; t < 2; ++t) {
            int tt = wave * 2 + t;
            int m0 = (tt >> 2) << 4, n0 = (tt & 3) << 4;
            v8f aq = {}, ak = {};
#pragma unroll
            for (int kk = 0; kk < 64; kk += 32) {
                v16h a   = ldA(hbuf, HSTR, m0, kk, lane);
                v16h b0  = ldB(wq_h, 64, n0, kk, lane);
                v16h b1f = ldB(wk_h, 64, n0, kk, lane);
                aq = wmma_f16(a, b0, aq);
                ak = wmma_f16(a, b1f, ak);
            }
            stD16(bufA, ASTR, m0, n0, lane, aq, 0.125f, bq + hd * 64);  // pre-scale 1/sqrt(64)
            stD16(bufB, ASTR, m0, n0, lane, ak, 1.0f,   bk + hd * 64);
        }
        __syncthreads();

        // ---- scores = q' x k^T (row-major k buffer IS B^T) ----
#pragma unroll
        for (int t = 0; t < 2; ++t) {
            int tt = wave * 2 + t;
            int m0 = (tt >> 2) << 4, n0 = (tt & 3) << 4;
            v8f s = {};
#pragma unroll
            for (int kk = 0; kk < 64; kk += 32)
                s = wmma_f16(ldA(bufA, ASTR, m0, kk, lane), ldB(bufB, ASTR, n0, kk, lane), s);
            stD32(sbuf, SSTR, m0, n0, lane, s);
        }
        __syncthreads();

        // ---- softmax: 4 lanes per row, 16 cols each -> P into bufA ----
        {
            int r = tid >> 2, p = tid & 3;
            const float4* row = (const float4*)(sbuf + r * SSTR + p * 16);
            float4 qa[4];
#pragma unroll
            for (int j = 0; j < 4; ++j) qa[j] = row[j];
            float mx = -1e30f;
#pragma unroll
            for (int j = 0; j < 4; ++j)
                mx = fmaxf(mx, fmaxf(fmaxf(qa[j].x, qa[j].y), fmaxf(qa[j].z, qa[j].w)));
            mx = red4_max(mx);
            float e[16]; float sum = 0.f;
#pragma unroll
            for (int j = 0; j < 4; ++j) {
                e[4 * j + 0] = __expf(qa[j].x - mx);
                e[4 * j + 1] = __expf(qa[j].y - mx);
                e[4 * j + 2] = __expf(qa[j].z - mx);
                e[4 * j + 3] = __expf(qa[j].w - mx);
                sum += e[4 * j + 0] + e[4 * j + 1] + e[4 * j + 2] + e[4 * j + 3];
            }
            sum = red4_add(sum);
            float rinv = 1.f / sum;
            _Float16* pp = bufA + r * ASTR + p * 16;
#pragma unroll
            for (int i = 0; i < 16; ++i) pp[i] = (_Float16)(e[i] * rinv);
        }

        // ---- V projection, stored transposed (d x token) into bufB ----
#pragma unroll
        for (int t = 0; t < 2; ++t) {
            int tt = wave * 2 + t;
            int m0 = (tt >> 2) << 4, n0 = (tt & 3) << 4;
            v8f av = {};
#pragma unroll
            for (int kk = 0; kk < 64; kk += 32)
                av = wmma_f16(ldA(hbuf, HSTR, m0, kk, lane), ldB(wv_h, 64, n0, kk, lane), av);
            stD16T(bufB, ASTR, m0, n0, lane, av, bv + hd * 64);
        }
        __syncthreads();

        // ---- oh = P x V (v^T rows give B^T directly) ----
#pragma unroll
        for (int t = 0; t < 2; ++t) {
            int tt = wave * 2 + t;
            int m0 = (tt >> 2) << 4, n0 = (tt & 3) << 4;
            v8f o = {};
#pragma unroll
            for (int kk = 0; kk < 64; kk += 32)
                o = wmma_f16(ldA(bufA, ASTR, m0, kk, lane), ldB(bufB, ASTR, n0, kk, lane), o);
            stD16p(bufC, ASTR, m0, n0, lane, o);
        }
        __syncthreads();

        // ---- y += oh x Wo[head]  (accumulate across heads in registers) ----
        {
            int m0 = (yt0 >> 2) << 4, n0 = (yt0 & 3) << 4;
#pragma unroll
            for (int kk = 0; kk < 64; kk += 32)
                yacc0 = wmma_f16(ldA(bufC, ASTR, m0, kk, lane), ldB(wso + hd * 64, 256, n0, kk, lane), yacc0);
            m0 = (yt1 >> 2) << 4; n0 = (yt1 & 3) << 4;
#pragma unroll
            for (int kk = 0; kk < 64; kk += 32)
                yacc1 = wmma_f16(ldA(bufC, ASTR, m0, kk, lane), ldB(wso + hd * 64, 256, n0, kk, lane), yacc1);
        }
    }

    // ---- y = 2*(proj + bo) -> sbuf ----
    {
        int m0 = (yt0 >> 2) << 4, n0 = (yt0 & 3) << 4;
        int col = n0 + (lane & 15), mb = m0 + ((lane >> 4) << 3);
        float bb = bo[col];
#pragma unroll
        for (int r = 0; r < 8; ++r) sbuf[(mb + r) * SSTR + col] = 2.f * (yacc0[r] + bb);
        m0 = (yt1 >> 2) << 4; n0 = (yt1 & 3) << 4;
        col = n0 + (lane & 15); mb = m0 + ((lane >> 4) << 3);
        bb = bo[col];
#pragma unroll
        for (int r = 0; r < 8; ++r) sbuf[(mb + r) * SSTR + col] = 2.f * (yacc1[r] + bb);
    }
    __syncthreads();

    // ---- LN2: 4 lanes per token -> h2 (f16) in hbuf ----
    {
        int m = tid >> 2, p = tid & 3;
        const float4* row = (const float4*)(sbuf + m * SSTR + p * 16);
        float4 qa[4];
#pragma unroll
        for (int j = 0; j < 4; ++j) qa[j] = row[j];
        float s = 0.f, s2 = 0.f;
#pragma unroll
        for (int j = 0; j < 4; ++j) {
            s  += qa[j].x + qa[j].y + qa[j].z + qa[j].w;
            s2 += qa[j].x * qa[j].x + qa[j].y * qa[j].y + qa[j].z * qa[j].z + qa[j].w * qa[j].w;
        }
        s = red4_add(s); s2 = red4_add(s2);
        float mean = s * (1.f / 64.f);
        float var  = s2 * (1.f / 64.f) - mean * mean;
        float rstd = rsqrtf(var + 1e-5f);
        const float4* g4 = (const float4*)(ln2_g + p * 16);
        const float4* b4 = (const float4*)(ln2_b + p * 16);
        _Float16* hp = hbuf + m * HSTR + p * 16;
#pragma unroll
        for (int j = 0; j < 4; ++j) {
            float4 g = g4[j], b = b4[j], v = qa[j];
            hp[4 * j + 0] = (_Float16)((v.x - mean) * rstd * g.x + b.x);
            hp[4 * j + 1] = (_Float16)((v.y - mean) * rstd * g.y + b.y);
            hp[4 * j + 2] = (_Float16)((v.z - mean) * rstd * g.z + b.z);
            hp[4 * j + 3] = (_Float16)((v.w - mean) * rstd * g.w + b.w);
        }
    }
    __syncthreads();

    // ---- FFN1: 64x256, GELU(exact erf) -> gbuf (aliases bufA/B/C/sbuf) ----
#pragma unroll
    for (int t = 0; t < 8; ++t) {
        int tt = wave * 8 + t;
        int m0 = (tt >> 4) << 4, n0 = (tt & 15) << 4;
        v8f acc = {};
#pragma unroll
        for (int kk = 0; kk < 64; kk += 32)
            acc = wmma_f16(ldA(hbuf, HSTR, m0, kk, lane), ldB(wsw1, 64, n0, kk, lane), acc);
        int col = n0 + (lane & 15), mb = m0 + ((lane >> 4) << 3);
        float bb = b1[col];
#pragma unroll
        for (int r = 0; r < 8; ++r) {
            float v = acc[r] + bb;
            float g = 0.5f * v * (1.f + erff(v * 0.70710678118f));
            gbuf[(mb + r) * GSTR + col] = (_Float16)g;
        }
    }
    __syncthreads();

    // ---- FFN2: 64x64, K=256, + b2, scatter-store to output voxels ----
#pragma unroll
    for (int t = 0; t < 2; ++t) {
        int tt = wave * 2 + t;
        int m0 = (tt >> 2) << 4, n0 = (tt & 3) << 4;
        v8f acc = {};
#pragma unroll
        for (int kk = 0; kk < 256; kk += 32)
            acc = wmma_f16(ldA(gbuf, GSTR, m0, kk, lane), ldB(wsw2, 256, n0, kk, lane), acc);
        int col = n0 + (lane & 15), mb = m0 + ((lane >> 4) << 3);
        float bb = b2[col];
#pragma unroll
        for (int r = 0; r < 8; ++r) {
            int m = mb + r;
            int vox = baseVox + (m >> 4) * 4096 + ((m >> 2) & 3) * 64 + (m & 3);
            out[(size_t)vox * 64 + col] = acc[r] + bb;
        }
    }
}

extern "C" void kernel_launch(void* const* d_in, const int* in_sizes, int n_in,
                              void* d_out, int out_size, void* d_ws, size_t ws_size,
                              hipStream_t stream) {
    const float* x     = (const float*)d_in[0];
    const float* ln1_g = (const float*)d_in[1];
    const float* ln1_b = (const float*)d_in[2];
    const float* wq    = (const float*)d_in[3];
    const float* bq    = (const float*)d_in[4];
    const float* wk    = (const float*)d_in[5];
    const float* bk    = (const float*)d_in[6];
    const float* wv    = (const float*)d_in[7];
    const float* bv    = (const float*)d_in[8];
    const float* wo    = (const float*)d_in[9];
    const float* bo    = (const float*)d_in[10];
    const float* ln2_g = (const float*)d_in[11];
    const float* ln2_b = (const float*)d_in[12];
    const float* w1    = (const float*)d_in[13];
    const float* b1    = (const float*)d_in[14];
    const float* w2    = (const float*)d_in[15];
    const float* b2    = (const float*)d_in[16];
    _Float16* ws = (_Float16*)d_ws;
    float* out = (float*)d_out;

    prep_weights<<<384, 256, 0, stream>>>(wq, wk, wv, wo, w1, w2, ws);
    swin_block<<<4096, 256, 0, stream>>>(x, ln1_g, ln1_b, bq, bk, bv, bo,
                                         ln2_g, ln2_b, b1, b2, ws, out);
}
